// TemporalSSM_2173253452152
// MI455X (gfx1250) — compile-verified, matrix operand order checked by reading, not run on hardware
//
#include <hip/hip_runtime.h>
#include <hip/hip_bf16.h>
#include <math.h>

// MI455X / gfx1250, wave32. Compile-only environment.
//
// TemporalSSM: per-pixel time series (B=2,T=15,H=192,W=192), C=P=16.
// One wave32 handles 16 consecutive sequences (16 | W so tiles are contiguous).
// All 16x16 channel/state matmuls -> v_wmma_f32_16x16x32_f16 (f32 accum).
// Only the last timestep feeds the output head, so the post-SSM stack runs once.
// x values for all T are preloaded into registers so the serial recurrence
// chain never waits on global memory.

typedef _Float16 v16h __attribute__((ext_vector_type(16)));
typedef _Float16 v8h  __attribute__((ext_vector_type(8)));
typedef float    v8f  __attribute__((ext_vector_type(8)));
typedef float    v4f  __attribute__((ext_vector_type(4)));

#define NB 2
#define NT 15
#define NH 192
#define NWD 192
#define NC 16
#define NP 16
#define HW (NH * NWD)
#define NSEQ (NB * HW)
#define WAVES_PER_BLOCK 8
#define SEQ_PER_WAVE 16

__device__ __forceinline__ float gelu_exact(float x) {
  return 0.5f * x * (1.0f + erff(x * 0.70710678118654752f));
}

__device__ __forceinline__ v8f wmma_f16(v16h a, v16h b, v8f c) {
  // (neg_a, A, neg_b, B, c_mod, C, reuse_a, reuse_b)
  return __builtin_amdgcn_wmma_f32_16x16x32_f16(false, a, false, b, (short)0, c,
                                                false, false);
}

// A-operand gather from row-major [16][32] f16 LDS tile:
// lane l holds row m=l%16; lanes 0-15 take K=0..7 & 16..23, lanes 16-31 take
// K=8..15 & 24..31 (two ds_load_b128 per lane).
__device__ __forceinline__ v16h load_A_tile(const _Float16* base, int m, int hi) {
  v8h lo = *(const v8h*)(base + m * 32 + hi * 8);
  v8h hh = *(const v8h*)(base + m * 32 + 16 + hi * 8);
  v16h a;
#pragma unroll
  for (int i = 0; i < 8; ++i) { a[i] = lo[i]; a[8 + i] = hh[i]; }
  return a;
}

// LayerNorm over the 16 accumulator columns (channels live in lane%16):
// shfl_xor masks 1,2,4,8 stay inside each 16-lane half under wave32.
__device__ __forceinline__ void ln_lanes(float* val, float gam, float bet) {
#pragma unroll
  for (int v = 0; v < 8; ++v) {
    float s = val[v], s2 = val[v] * val[v];
#pragma unroll
    for (int msk = 1; msk < 16; msk <<= 1) {
      s  += __shfl_xor(s,  msk, 32);
      s2 += __shfl_xor(s2, msk, 32);
    }
    float mu  = s * (1.0f / 16.0f);
    float var = s2 * (1.0f / 16.0f) - mu * mu;
    val[v] = (val[v] - mu) * rsqrtf(var + 1e-5f) * gam + bet;
  }
}

__global__ __launch_bounds__(256) void temporal_ssm_kernel(
    const float* __restrict__ x,
    const float* __restrict__ w_in,      const float* __restrict__ b_in,
    const float* __restrict__ attn_g,    const float* __restrict__ attn_b,
    const float* __restrict__ Lam_re,    const float* __restrict__ Lam_im,
    const float* __restrict__ Bre,       const float* __restrict__ Bim,
    const float* __restrict__ Cre,       const float* __restrict__ Cim,
    const float* __restrict__ Dvec,      const float* __restrict__ log_step,
    const float* __restrict__ ff_g,      const float* __restrict__ ff_b,
    const float* __restrict__ Wenc,      const float* __restrict__ Wdec,
    const float* __restrict__ out_g,     const float* __restrict__ out_b,
    const float* __restrict__ Wout,      const float* __restrict__ bout,
    float* __restrict__ y_out)
{
  __shared__ _Float16 astage[WAVES_PER_BLOCK][16][32];  // 8 KB total

  const int tid  = threadIdx.x;
  const int lane = tid & 31;
  const int wav  = tid >> 5;
  const int hi   = lane >> 4;    // 0/1 half-wave select
  const int nsel = lane & 15;    // accumulator column / A row

  const int gwave = blockIdx.x * WAVES_PER_BLOCK + wav;
  const int seq0  = gwave * SEQ_PER_WAVE;
  if (seq0 >= NSEQ) return;      // exact launch; never divergent

  const int bb  = seq0 / HW;
  const int rem = seq0 - bb * HW;
  const int hh  = rem / NWD;
  const int w0  = rem - hh * NWD;

  _Float16* my = &astage[wav][0][0];

  // ---- preload the whole time series for this lane's sequence ----------
  // 15 independent dword loads issued back-to-back (coalesced 64B per wave
  // per t, non-temporal: x is streamed exactly once).
  const float* xbase = x + (size_t)bb * NT * HW + (size_t)hh * NWD + w0 + nsel;
  float xv[NT];
#pragma unroll
  for (int t = 0; t < NT; ++t)
    xv[t] = __builtin_nontemporal_load(xbase + (size_t)t * HW);

  // ---------------- per-lane derived constants -------------------------
  // This lane's state p = nsel: discretization Lam_bar = exp(Lam*step),
  // coef = (Lam_bar - 1)/Lam (complex).
  const int p = nsel;
  float lre = Lam_re[p], lim = Lam_im[p];
  float stp = expf(log_step[p]);
  float emag = expf(lre * stp);
  float lbr = emag * cosf(lim * stp);
  float lbi = emag * sinf(lim * stp);
  float den = lre * lre + lim * lim;
  float cr  = ((lbr - 1.0f) * lre + lbi * lim) / den;
  float ci  = (lbi * lre - (lbr - 1.0f) * lim) / den;

  // B operands held in registers for the whole kernel (lane = output col nsel,
  // element e -> K = hi*8+e, element 8+e -> K = 16+hi*8+e).
  v16h B_bur, B_bui, B_y, B_e0, B_e1, B_dc, B_ot;
#pragma unroll
  for (int e = 0; e < 8; ++e) {
    int k = hi * 8 + e;  // low-K index in [0,16)
    // Bu projection:  Bu[s,p] = sum_c fx[s,c] * Bbar[p,c]  (n=p, k=c)
    float br = Bre[p * NC + k], bi = Bim[p * NC + k];
    B_bur[e] = (_Float16)(cr * br - ci * bi);  B_bur[8 + e] = (_Float16)0.0f;
    B_bui[e] = (_Float16)(cr * bi + ci * br);  B_bui[8 + e] = (_Float16)0.0f;
    // y projection: K=32 packs [xs_re | xs_im] against [C_re ; -C_im] (masked)
    {
      float fr  = expf(log_step[k]) * fabsf(Lam_im[k]) * 0.15915494309f;
      float msk = (fr < 0.25f) ? 1.0f : 0.0f;
      B_y[e]     = (_Float16)( Cre[nsel * NP + k] * msk);
      B_y[8 + e] = (_Float16)(-Cim[nsel * NP + k] * msk);
    }
    // encoder halves, decoder, output head: n = out channel, k = in channel
    B_e0[e] = (_Float16)Wenc[nsel * NC + k];         B_e0[8 + e] = (_Float16)0.0f;
    B_e1[e] = (_Float16)Wenc[(nsel + NC) * NC + k];  B_e1[8 + e] = (_Float16)0.0f;
    B_dc[e] = (_Float16)Wdec[nsel * NC + k];         B_dc[8 + e] = (_Float16)0.0f;
    B_ot[e] = (_Float16)Wout[nsel * NC + k];         B_ot[8 + e] = (_Float16)0.0f;
  }

  // per-lane channel constants for A-side fx (8 channels c = hi*8+e)
  float w8[8], bi8[8], g8[8], be8[8];
#pragma unroll
  for (int e = 0; e < 8; ++e) {
    int c = hi * 8 + e;
    w8[e] = w_in[c]; bi8[e] = b_in[c]; g8[e] = attn_g[c]; be8[e] = attn_b[c];
  }
  // D-layout channel constants (c = nsel)
  float winc = w_in[nsel], binc = b_in[nsel], agc = attn_g[nsel], abc = attn_b[nsel];
  float Dc = Dvec[nsel], fgc = ff_g[nsel], fbc = ff_b[nsel];
  float ogc = out_g[nsel], obc = out_b[nsel], boc = bout[nsel];

  // Input LN collapses to closed form: u_c = x*w_c + b_c is affine in scalar x,
  // so mean/var need only the 5 moments of (w_in, b_in).
  float Mw = 0, Mb = 0, Mww = 0, Mwb = 0, Mbb = 0;
  for (int j = 0; j < NC; ++j) {
    float wj = w_in[j], bj = b_in[j];
    Mw += wj; Mb += bj; Mww += wj * wj; Mwb += wj * bj; Mbb += bj * bj;
  }
  Mw *= 0.0625f; Mb *= 0.0625f; Mww *= 0.0625f; Mwb *= 0.0625f; Mbb *= 0.0625f;

  // ---------------- SSM recurrence loop over T -------------------------
  // Accumulator layout: element v, lane l  <->  (seq s = v + 8*hi, col = nsel).
  // States live in the column dim, so Lam_bar is a per-lane scalar and the
  // recurrence is pure elementwise VALU between WMMAs. Fully unrolled:
  // no memory ops on the serial chain.
  v8f xs_re = {}; v8f xs_im = {};
#pragma unroll
  for (int t = 0; t < NT; ++t) {
    float xt = xv[t];
    float mu = xt * Mw + Mb;
    float m2 = xt * xt * Mww + 2.0f * xt * Mwb + Mbb;
    float rstd = rsqrtf(fmaxf(m2 - mu * mu, 0.0f) + 1e-5f);
    v16h a;                                          // fx built in A layout
#pragma unroll
    for (int e = 0; e < 8; ++e) {
      float u = xt * w8[e] + bi8[e];
      a[e]     = (_Float16)((u - mu) * rstd * g8[e] + be8[e]);
      a[8 + e] = (_Float16)0.0f;                     // pad K 16..31
    }
    v8f zc = {};
    v8f bur = wmma_f16(a, B_bur, zc);
    v8f bui = wmma_f16(a, B_bui, zc);
#pragma unroll
    for (int v = 0; v < 8; ++v) {
      float r = xs_re[v], i = xs_im[v];
      xs_re[v] = lbr * r - lbi * i + bur[v];
      xs_im[v] = lbr * i + lbi * r + bui[v];
    }
  }

  // ---------------- tail: only t = T-1 feeds the head ------------------
  // fx(T-1) in D layout (needs x of row s -> ds_bpermute broadcast)
  float lastx = xv[NT - 1];
  float fxD[8];
#pragma unroll
  for (int v = 0; v < 8; ++v) {
    float xs_ = __shfl(lastx, v + 8 * hi, 32);
    float mu = xs_ * Mw + Mb;
    float m2 = xs_ * xs_ * Mww + 2.0f * xs_ * Mwb + Mbb;
    float rstd = rsqrtf(fmaxf(m2 - mu * mu, 0.0f) + 1e-5f);
    fxD[v] = (xs_ * winc + binc - mu) * rstd * agc + abc;
  }

  // stage xs (re -> K 0..15, im -> K 16..31) and project: y = Re(xs @ Cm^T)
#pragma unroll
  for (int v = 0; v < 8; ++v) {
    int s = v + 8 * hi;
    my[s * 32 + nsel]      = (_Float16)xs_re[v];
    my[s * 32 + 16 + nsel] = (_Float16)xs_im[v];
  }
  asm volatile("s_wait_dscnt 0" ::: "memory");
  v8f zc = {};
  v16h axs = load_A_tile(my, nsel, hi);
  v8f yv = wmma_f16(axs, B_y, zc);

  float zarr[8];
#pragma unroll
  for (int v = 0; v < 8; ++v) {
    float yy = yv[v] + fxD[v] * Dc;          // + u*D
    zarr[v] = gelu_exact(yy) + fxD[v];       // z = gelu(s5) + fx
  }
  ln_lanes(zarr, fgc, fbc);                  // fx2

  // re-zero K 16..31 of the staging tile once (covers enc/dec/out stages)
  { v8h zz = {}; *(v8h*)(my + (lane >> 1) * 32 + 16 + (lane & 1) * 8) = zz; }

  // encoder: fx2 @ W_enc^T  (two N=16 halves, shared A)
#pragma unroll
  for (int v = 0; v < 8; ++v) my[(v + 8 * hi) * 32 + nsel] = (_Float16)zarr[v];
  asm volatile("s_wait_dscnt 0" ::: "memory");
  v16h af2 = load_A_tile(my, nsel, hi);
  v8f ea = wmma_f16(af2, B_e0, zc);
  v8f eg = wmma_f16(af2, B_e1, zc);

  float hg[8];
#pragma unroll
  for (int v = 0; v < 8; ++v) hg[v] = ea[v] * gelu_exact(eg[v]);

  // decoder: (a*gelu(g)) @ W_dec^T + fx2
#pragma unroll
  for (int v = 0; v < 8; ++v) my[(v + 8 * hi) * 32 + nsel] = (_Float16)hg[v];
  asm volatile("s_wait_dscnt 0" ::: "memory");
  v16h ah = load_A_tile(my, nsel, hi);
  v8f dc = wmma_f16(ah, B_dc, zc);
  float z2[8];
#pragma unroll
  for (int v = 0; v < 8; ++v) z2[v] = dc[v] + zarr[v];

  // output head: LN(z2) @ W_out^T + b_out
  ln_lanes(z2, ogc, obc);
#pragma unroll
  for (int v = 0; v < 8; ++v) my[(v + 8 * hi) * 32 + nsel] = (_Float16)z2[v];
  asm volatile("s_wait_dscnt 0" ::: "memory");
  v16h ao = load_A_tile(my, nsel, hi);
  v8f ov = wmma_f16(ao, B_ot, zc);

  // store y[b][c][h][w]: lane owns channel nsel, rows s = 8*hi .. 8*hi+7
  // (clang native v4f so the nontemporal builtin accepts it -> global_store_b128)
  float* obase = y_out + (((size_t)bb * NC + nsel) * NH + hh) * NWD + w0 + 8 * hi;
  v4f o0 = { ov[0] + boc, ov[1] + boc, ov[2] + boc, ov[3] + boc };
  v4f o1 = { ov[4] + boc, ov[5] + boc, ov[6] + boc, ov[7] + boc };
  __builtin_nontemporal_store(o0, (v4f*)(obase));
  __builtin_nontemporal_store(o1, (v4f*)(obase + 4));
}

extern "C" void kernel_launch(void* const* d_in, const int* in_sizes, int n_in,
                              void* d_out, int out_size, void* d_ws, size_t ws_size,
                              hipStream_t stream) {
  (void)in_sizes; (void)n_in; (void)out_size; (void)d_ws; (void)ws_size;
  const float* x        = (const float*)d_in[0];
  const float* w_in     = (const float*)d_in[1];
  const float* b_in     = (const float*)d_in[2];
  const float* attn_g   = (const float*)d_in[3];
  const float* attn_b   = (const float*)d_in[4];
  const float* Lam_re   = (const float*)d_in[5];
  const float* Lam_im   = (const float*)d_in[6];
  const float* Bre      = (const float*)d_in[7];
  const float* Bim      = (const float*)d_in[8];
  const float* Cre      = (const float*)d_in[9];
  const float* Cim      = (const float*)d_in[10];
  const float* Dvec     = (const float*)d_in[11];
  const float* log_step = (const float*)d_in[12];
  const float* ff_g     = (const float*)d_in[13];
  const float* ff_b     = (const float*)d_in[14];
  const float* Wenc     = (const float*)d_in[15];
  const float* Wdec     = (const float*)d_in[16];
  const float* out_g    = (const float*)d_in[17];
  const float* out_b    = (const float*)d_in[18];
  const float* Wout     = (const float*)d_in[19];
  const float* bout     = (const float*)d_in[20];

  const int total_waves = NSEQ / SEQ_PER_WAVE;           // 4608
  const int blocks      = total_waves / WAVES_PER_BLOCK; // 576
  temporal_ssm_kernel<<<blocks, WAVES_PER_BLOCK * 32, 0, stream>>>(
      x, w_in, b_in, attn_g, attn_b, Lam_re, Lam_im, Bre, Bim, Cre, Cim,
      Dvec, log_step, ff_g, ff_b, Wenc, Wdec, out_g, out_b, Wout, bout,
      (float*)d_out);
}
